// ArcMarginProduct_59906203845142
// MI455X (gfx1250) — compile-verified, matrix operand order checked by reading, not run on hardware
//
#include <hip/hip_runtime.h>
#include <stdint.h>

#define IN_F   512
#define OUT_F  11014
#define NROWS  8192
#define CPAD   11136          // 87 * 128, padded class count for tiling
#define LDSSTR 40             // bf16 elems per LDS tile row (32 data + 8 pad, 80B: 16B aligned, conflict-free)
#define KCH    (IN_F / 32)    // 16 K-chunks of 32

typedef __attribute__((ext_vector_type(16))) __bf16 v16bf;
typedef __attribute__((ext_vector_type(8)))  __bf16 v8bf;
typedef __attribute__((ext_vector_type(8)))  float  v8f;

// ArcFace constants (m = 0.6, s = 30)
#define ARC_S    30.0f
#define ARC_COSM 0.8253356149096783f
#define ARC_SINM 0.5646424733950354f
#define ARC_TH  (-0.8253356149096783f)          /* cos(pi - m) */
#define ARC_MM   0.33878548403702124f           /* sin(pi - m) * m */

__device__ __forceinline__ unsigned short f2bf(float f) {
    unsigned u = __float_as_uint(f);
    u = u + 0x7FFFu + ((u >> 16) & 1u);         // round-to-nearest-even
    return (unsigned short)(u >> 16);
}

// ---------------------------------------------------------------------------
// gfx1250 async copy: global -> LDS, 16B per lane, tracked by ASYNCcnt.
// VDST operand = LDS byte offset (low 32 bits of generic pointer to __shared__).
// ---------------------------------------------------------------------------
__device__ __forceinline__ void async_b128(const unsigned short* gsrc, unsigned lds_off) {
    unsigned long long ga = (unsigned long long)(uintptr_t)gsrc;
    asm volatile("global_load_async_to_lds_b128 %0, %1, off"
                 :: "v"(lds_off), "v"(ga)
                 : "memory");
}

__device__ __forceinline__ void wait_async0() {
#if __has_builtin(__builtin_amdgcn_s_wait_asynccnt)
    __builtin_amdgcn_s_wait_asynccnt(0);
#else
    asm volatile("s_wait_asynccnt 0x0" ::: "memory");
#endif
}

// ---------------------------------------------------------------------------
// Row L2-normalize f32 -> bf16.  One wave per row (512 floats, 16 per lane).
// Rows in [rows, rows_pad) are zero-filled padding (safe WMMA operands).
// ---------------------------------------------------------------------------
__global__ __launch_bounds__(256)
void normalize_rows(const float* __restrict__ src, unsigned short* __restrict__ dst,
                    int rows, int rows_pad) {
    const int wave = threadIdx.x >> 5;
    const int lane = threadIdx.x & 31;
    const int r = blockIdx.x * 8 + wave;
    if (r >= rows_pad) return;

    if (r >= rows) {
        uint2 z; z.x = 0u; z.y = 0u;
        #pragma unroll
        for (int i = 0; i < 4; ++i)
            *reinterpret_cast<uint2*>(dst + (size_t)r * IN_F + i * 128 + lane * 4) = z;
        return;
    }

    const float4* p = reinterpret_cast<const float4*>(src + (size_t)r * IN_F);
    float4 v[4];
    float s = 0.f;
    #pragma unroll
    for (int i = 0; i < 4; ++i) {
        v[i] = p[i * 32 + lane];
        s += v[i].x * v[i].x + v[i].y * v[i].y + v[i].z * v[i].z + v[i].w * v[i].w;
    }
    #pragma unroll
    for (int m = 16; m >= 1; m >>= 1) s += __shfl_xor(s, m, 32);
    const float inv = 1.0f / fmaxf(sqrtf(s), 1e-12f);

    #pragma unroll
    for (int i = 0; i < 4; ++i) {
        uint2 w;
        w.x = (unsigned)f2bf(v[i].x * inv) | ((unsigned)f2bf(v[i].y * inv) << 16);
        w.y = (unsigned)f2bf(v[i].z * inv) | ((unsigned)f2bf(v[i].w * inv) << 16);
        *reinterpret_cast<uint2*>(dst + (size_t)r * IN_F + i * 128 + lane * 4) = w;
    }
}

// ---------------------------------------------------------------------------
// Fused cosine-GEMM + ArcFace margin epilogue.
// 128x128 block tile, 8 waves (4M x 2N), each wave 2x4 WMMA 16x16 tiles.
// Double-buffered LDS fed by async global->LDS copies (ASYNCcnt).
// ---------------------------------------------------------------------------
__device__ __forceinline__ v16bf load_frag(const unsigned short* base) {
    v8bf lo = *reinterpret_cast<const v8bf*>(base);        // K run h*8 .. h*8+7
    v8bf hi = *reinterpret_cast<const v8bf*>(base + 16);   // K run 16+h*8 .. +7
    return __builtin_shufflevector(lo, hi, 0, 1, 2, 3, 4, 5, 6, 7,
                                            8, 9, 10, 11, 12, 13, 14, 15);
}

__global__ __launch_bounds__(256)
void arcface_gemm(const unsigned short* __restrict__ xn,
                  const unsigned short* __restrict__ wn,
                  const int* __restrict__ label,
                  float* __restrict__ out) {
    __shared__ __align__(16) unsigned short ldsA[2][128 * LDSSTR];
    __shared__ __align__(16) unsigned short ldsB[2][128 * LDSSTR];
    __shared__ int ldsLab[128];

    const int tid  = threadIdx.x;
    const int lane = tid & 31;
    const int wave = tid >> 5;
    const int wm   = wave & 3;          // wave M index: rows wm*32 .. +31
    const int wnn  = wave >> 2;         // wave N index: cols wnn*64 .. +63
    const int rowBase = blockIdx.y * 128;
    const int colBase = blockIdx.x * 128;

    if (tid < 128) ldsLab[tid] = label[rowBase + tid];

    // Per-thread staging slots: 2 x 16B per matrix per chunk (128x32 bf16 tile).
    const int lrow = tid >> 1;                    // 0..127
    const int lg0  = (tid & 1) * 2;               // groups {0,1} or {2,3}
    const size_t gA = (size_t)(rowBase + lrow) * IN_F;
    const size_t gB = (size_t)(colBase + lrow) * IN_F;

    v8f acc[2][4];
    #pragma unroll
    for (int i = 0; i < 2; ++i)
        #pragma unroll
        for (int j = 0; j < 4; ++j) {
            v8f z = {};
            acc[i][j] = z;
        }

    const int h  = lane >> 4;           // lane half
    const int ml = lane & 15;           // row/col within 16x16 tile

    // Prologue: async-stage chunk 0 into buffer 0.
    #pragma unroll
    for (int it = 0; it < 2; ++it) {
        const int g = lg0 + it;
        async_b128(xn + gA + g * 8, (unsigned)(uintptr_t)&ldsA[0][lrow * LDSSTR + g * 8]);
        async_b128(wn + gB + g * 8, (unsigned)(uintptr_t)&ldsB[0][lrow * LDSSTR + g * 8]);
    }
    wait_async0();
    __syncthreads();

    for (int kc = 0; kc < KCH; ++kc) {
        const int cur = kc & 1;
        const int nxt = cur ^ 1;

        // Kick off async DMA of the next K-chunk into the other buffer; it
        // overlaps this chunk's fragment loads + WMMAs.
        if (kc + 1 < KCH) {
            const int kk = (kc + 1) * 32;
            #pragma unroll
            for (int it = 0; it < 2; ++it) {
                const int g = lg0 + it;
                async_b128(xn + gA + kk + g * 8,
                           (unsigned)(uintptr_t)&ldsA[nxt][lrow * LDSSTR + g * 8]);
                async_b128(wn + gB + kk + g * 8,
                           (unsigned)(uintptr_t)&ldsB[nxt][lrow * LDSSTR + g * 8]);
            }
        }

        v16bf afr[2], bfr[4];
        #pragma unroll
        for (int i = 0; i < 2; ++i)
            afr[i] = load_frag(&ldsA[cur][(wm * 32 + i * 16 + ml) * LDSSTR + h * 8]);
        #pragma unroll
        for (int j = 0; j < 4; ++j)
            bfr[j] = load_frag(&ldsB[cur][(wnn * 64 + j * 16 + ml) * LDSSTR + h * 8]);

        #pragma unroll
        for (int i = 0; i < 2; ++i)
            #pragma unroll
            for (int j = 0; j < 4; ++j)
                acc[i][j] = __builtin_amdgcn_wmma_f32_16x16x32_bf16(
                    false, afr[i], false, bfr[j], (short)0, acc[i][j], false, false);

        // DMA for chunk kc+1 must have landed, and all waves must be done
        // reading buffer `cur`, before the next iteration touches them.
        wait_async0();
        __syncthreads();
    }

    // Fused ArcFace margin epilogue: out = s * (onehot ? phi : cos)
    #pragma unroll
    for (int i = 0; i < 2; ++i) {
        #pragma unroll
        for (int j = 0; j < 4; ++j) {
            const int colG  = colBase + wnn * 64 + j * 16 + ml;
            const bool colOK = colG < OUT_F;
            #pragma unroll
            for (int e = 0; e < 8; ++e) {
                const int rowL = wm * 32 + i * 16 + e + h * 8;   // C layout: M = e + 8*h
                float c  = acc[i][j][e];
                float s2 = fmaxf(1.0f - c * c, 0.0f);
                float ph = c * ARC_COSM - sqrtf(s2) * ARC_SINM;
                ph = (c > ARC_TH) ? ph : (c - ARC_MM);
                float val = (ldsLab[rowL] == colG) ? ph : c;
                if (colOK)
                    out[(size_t)(rowBase + rowL) * OUT_F + colG] = val * ARC_S;
            }
        }
    }
}

// ---------------------------------------------------------------------------
// Per-row online logsumexp -> row loss = lse(row) - logit[label].
// ---------------------------------------------------------------------------
__global__ __launch_bounds__(256)
void row_lse(const float* __restrict__ out, const int* __restrict__ label,
             float* __restrict__ rowloss) {
    __shared__ float sm[256];
    __shared__ float ss[256];
    const int tid = threadIdx.x;
    const int r = blockIdx.x;
    const float* o = out + (size_t)r * OUT_F;

    float m = -3.4e38f, s = 0.0f;
    for (int c = tid; c < OUT_F; c += 256) {
        float v = o[c];
        if (v > m) { s = s * __expf(m - v) + 1.0f; m = v; }
        else       { s += __expf(v - m); }
    }
    sm[tid] = m; ss[tid] = s;
    __syncthreads();
    for (int off = 128; off >= 1; off >>= 1) {
        if (tid < off) {
            float m2 = sm[tid + off], s2 = ss[tid + off];
            float mm = fmaxf(sm[tid], m2);
            ss[tid] = ss[tid] * __expf(sm[tid] - mm) + s2 * __expf(m2 - mm);
            sm[tid] = mm;
        }
        __syncthreads();
    }
    if (tid == 0)
        rowloss[r] = (sm[0] + logf(ss[0])) - o[label[r]];
}

__global__ __launch_bounds__(256)
void mean_loss(const float* __restrict__ rowloss, float* __restrict__ loss_out) {
    __shared__ float sm[256];
    float s = 0.0f;
    for (int i = threadIdx.x; i < NROWS; i += 256) s += rowloss[i];
    sm[threadIdx.x] = s;
    __syncthreads();
    for (int off = 128; off >= 1; off >>= 1) {
        if (threadIdx.x < off) sm[threadIdx.x] += sm[threadIdx.x + off];
        __syncthreads();
    }
    if (threadIdx.x == 0) *loss_out = sm[0] / (float)NROWS;
}

// ---------------------------------------------------------------------------
extern "C" void kernel_launch(void* const* d_in, const int* in_sizes, int n_in,
                              void* d_out, int out_size, void* d_ws, size_t ws_size,
                              hipStream_t stream) {
    const float* x     = (const float*)d_in[0];
    const int*   label = (const int*)d_in[1];
    const float* w     = (const float*)d_in[2];
    float* out = (float*)d_out;

    unsigned short* xn = (unsigned short*)d_ws;                 // 8192*512 bf16  (8 MB)
    unsigned short* wn = xn + (size_t)NROWS * IN_F;             // 11136*512 bf16 (~11.4 MB)
    float* rowloss = (float*)(wn + (size_t)CPAD * IN_F);        // 8192 f32

    normalize_rows<<<NROWS / 8, 256, 0, stream>>>(x, xn, NROWS, NROWS);
    normalize_rows<<<CPAD / 8, 256, 0, stream>>>(w, wn, OUT_F, CPAD);

    dim3 grid(CPAD / 128, NROWS / 128);                         // 87 x 64
    arcface_gemm<<<grid, 256, 0, stream>>>(xn, wn, label, out);

    row_lse<<<NROWS, 256, 0, stream>>>(out, label, rowloss);
    mean_loss<<<1, 256, 0, stream>>>(rowloss, out + (size_t)NROWS * OUT_F);
}
